// TransformerBlock_36301063586009
// MI455X (gfx1250) — matrix operand order repaired and checked by reading, not compiled, not run on hardware
//
#include <hip/hip_runtime.h>
#include <hip/hip_bf16.h>

typedef __attribute__((ext_vector_type(16))) __bf16 v16bf;
typedef __attribute__((ext_vector_type(8)))  __bf16 v8bf;
typedef __attribute__((ext_vector_type(8)))  float  v8f;
typedef __attribute__((ext_vector_type(4)))  float  v4f;

static constexpr int Bb = 2, Tt = 4, Cch = 64, Himg = 128, Wimg = 128;
static constexpr int BT = Bb * Tt;            // 8
static constexpr int HW = Himg * Wimg;        // 16384
static constexpr int HEADS = 4, HD = 16, WIN = 8;
static constexpr int HIDc = 170;              // int(64*2.66)
static constexpr size_t NP = (size_t)BT * HW; // 131072 pixels

__device__ inline __bf16 f2bf(float f) {
    unsigned u = __float_as_uint(f);
    unsigned r = (u + 0x7FFFu + ((u >> 16) & 1u)) >> 16;
    unsigned short s = (unsigned short)r;
    __bf16 b;
    __builtin_memcpy(&b, &s, 2);
    return b;
}

__device__ inline v16bf cat8(v8bf lo, v8bf hi) {
    return __builtin_shufflevector(lo, hi, 0, 1, 2, 3, 4, 5, 6, 7,
                                   8, 9, 10, 11, 12, 13, 14, 15);
}

// ---------------------------------------------------------------- LayerNorm
__global__ __launch_bounds__(256) void ln_kernel(const float* __restrict__ in,
                                                 const float* __restrict__ nw,
                                                 const float* __restrict__ nb,
                                                 float* __restrict__ out) {
    size_t p = (size_t)blockIdx.x * 256 + threadIdx.x;
    int bt = (int)(p / HW); int pos = (int)(p % HW);
    const float* base = in + ((size_t)bt * Cch) * HW + pos;
    float s = 0.f, s2 = 0.f;
#pragma unroll 8
    for (int c = 0; c < Cch; ++c) { float v = base[(size_t)c * HW]; s += v; s2 += v * v; }
    float mu = s * (1.f / Cch);
    float var = s2 * (1.f / Cch) - mu * mu;
    float inv = rsqrtf(var + 1e-5f);
    float* ob = out + ((size_t)bt * Cch) * HW + pos;
#pragma unroll 8
    for (int c = 0; c < Cch; ++c) {
        float v = base[(size_t)c * HW];
        ob[(size_t)c * HW] = (v - mu) * inv * nw[c] + nb[c];
    }
}

// ---------------------------- pack weights to zero-padded bf16 [MT16][Kpad]
__global__ __launch_bounds__(256) void pack_w_kernel(const float* __restrict__ w,
                                                     __bf16* __restrict__ wp,
                                                     int Cout, int Kdim, int Kpad) {
    int MT16 = ((Cout + 15) >> 4) << 4;
    int total = MT16 * Kpad;
    int idx = blockIdx.x * 256 + threadIdx.x;
    if (idx >= total) return;
    int oc = idx / Kpad, k = idx % Kpad;
    float v = (oc < Cout && k < Kdim) ? w[(size_t)oc * Kdim + k] : 0.f;
    wp[idx] = f2bf(v);
}

// ------------------------------------------- generic (temporal/1x1) conv GEMM
// A = packed weights (global bf16), B = activation tile in LDS [pixel][K].
template <bool TEMPORAL>
__global__ __launch_bounds__(256) void gemm_conv_kernel(
    const float* __restrict__ in, const __bf16* __restrict__ wp,
    const float* __restrict__ bias, const float* __restrict__ res,
    float* __restrict__ out, int Cin, int Cout, int Kdim, int Kpad) {
    __shared__ __attribute__((aligned(32))) __bf16 sAct[64][224]; // [pixel][K]
    int tid = threadIdx.x;
    int P0 = blockIdx.x * 64;
    int bt = P0 / HW, pix0 = P0 % HW;
    int b = bt / Tt, t = bt % Tt;
    for (int idx = tid; idx < Kpad * 64; idx += 256) {
        int k = idx >> 6, p = idx & 63;            // p fastest -> coalesced
        float v = 0.f;
        if (k < Kdim) {
            int ic, tsrc;
            if (TEMPORAL) { ic = k / 3; tsrc = t + (k % 3) - 1; }
            else          { ic = k;     tsrc = t; }
            if (tsrc >= 0 && tsrc < Tt)
                v = in[((size_t)(b * Tt + tsrc) * Cin + ic) * HW + pix0 + p];
        }
        sAct[p][k] = f2bf(v);
    }
    __syncthreads();
    int wave = tid >> 5, lane = tid & 31;
    int ntile = wave & 3, mstart = wave >> 2;
    int m = lane & 15, hi = (lane >> 4) & 1;
    int mofs = hi * 8, n = lane & 15;
    int MT = (Cout + 15) >> 4;
    const v8bf* brow = (const v8bf*)(&sAct[ntile * 16 + n][0]);
    for (int mt = mstart; mt < MT; mt += 2) {
        v8f c = {};
        int oc = mt * 16 + m;
        const v8bf* arow = (const v8bf*)(wp + (size_t)oc * Kpad);
        for (int k0 = 0; k0 < Kpad; k0 += 32) {
            int q = k0 >> 3;
            v16bf a = cat8(arow[q + hi], arow[q + 2 + hi]);
            v16bf bfr = cat8(brow[q + hi], brow[q + 2 + hi]);
            c = __builtin_amdgcn_wmma_f32_16x16x32_bf16(false, a, false, bfr,
                                                        (short)0, c, false, false);
        }
#pragma unroll
        for (int rr = 0; rr < 8; ++rr) {
            int ocs = mt * 16 + mofs + rr;
            if (ocs < Cout) {
                size_t o = ((size_t)bt * Cout + ocs) * HW + pix0 + ntile * 16 + n;
                float v = c[rr];
                if (bias) v += bias[ocs];
                if (res)  v += res[o];
                out[o] = v;
            }
        }
    }
}

// ------------------------------------------------------- depthwise 3x3 conv
__global__ __launch_bounds__(256) void dw_kernel(const float* __restrict__ in,
                                                 const float* __restrict__ w,
                                                 const float* __restrict__ bias,
                                                 float* __restrict__ out,
                                                 int Ch, int dil) {
    size_t idx = (size_t)blockIdx.x * 256 + threadIdx.x;
    size_t total = (size_t)BT * Ch * HW;
    if (idx >= total) return;
    int pos = (int)(idx % HW);
    size_t t1 = idx / HW;
    int c = (int)(t1 % Ch); int bt = (int)(t1 / Ch);
    int hh = pos >> 7, ww = pos & 127;
    const float* base = in + ((size_t)bt * Ch + c) * HW;
    const float* wc = w + (size_t)c * 9;
    float acc = bias ? bias[c] : 0.f;
#pragma unroll
    for (int ky = 0; ky < 3; ++ky)
#pragma unroll
        for (int kx = 0; kx < 3; ++kx) {
            int y = hh + (ky - 1) * dil, x = ww + (kx - 1) * dil;
            if (y >= 0 && y < Himg && x >= 0 && x < Wimg)
                acc += wc[ky * 3 + kx] * base[y * Wimg + x];
        }
    out[idx] = acc;
}

// ------------------------------------------------ swin window attention (WMMA)
__global__ __launch_bounds__(256) void swsa_attn_kernel(const float* __restrict__ qkv,
                                                        const float* __restrict__ rpb,
                                                        float* __restrict__ out) {
    // per-head, K-padded Q/K: dims 16..31 pre-zeroed; V transposed [ch][tok]
    __shared__ __attribute__((aligned(32))) __bf16 sQ[HEADS][64][32];
    __shared__ __attribute__((aligned(32))) __bf16 sK[HEADS][64][32];
    __shared__ __attribute__((aligned(32))) __bf16 sVt[64][64];
    __shared__ float sA[HEADS][64][64];
    int tid = threadIdx.x;
    int wi = blockIdx.x;
    int bt = wi >> 8; int r0 = wi & 255;
    int wh0 = (r0 >> 4) << 3, ww0 = (r0 & 15) << 3;
    for (int idx = tid; idx < 64 * 64; idx += 256) {
        int tok = idx & 63, ch = idx >> 6;         // tok fastest -> coalesced-ish
        int h = ch >> 4, d = ch & 15;
        size_t g = ((size_t)bt * 192 + ch) * HW +
                   (size_t)(wh0 + (tok >> 3)) * Wimg + ww0 + (tok & 7);
        sQ[h][tok][d] = f2bf(qkv[g] * 0.25f);      // SCALE = hd^-0.5
        sK[h][tok][d] = f2bf(qkv[g + (size_t)64 * HW]);
        sVt[ch][tok]  = f2bf(qkv[g + (size_t)128 * HW]);
    }
    for (int idx = tid; idx < HEADS * 64 * 16; idx += 256) {
        int h = idx >> 10; int r = idx & 1023; int tok = r >> 4; int d = r & 15;
        sQ[h][tok][16 + d] = f2bf(0.f);
        sK[h][tok][16 + d] = f2bf(0.f);
    }
    __syncthreads();
    int wave = tid >> 5, lane = tid & 31;
    int head = wave >> 1;
    int m = lane & 15, hi = (lane >> 4) & 1;
    int kb8 = hi * 8, mofs = hi * 8, n = lane & 15;
    // phase 1: attn = Q K^T  (K = 16 padded to 32, zeros baked in LDS)
    for (int mt = (wave & 1) * 2; mt < (wave & 1) * 2 + 2; ++mt) {
        const v8bf* qrow = (const v8bf*)(&sQ[head][mt * 16 + m][0]);
        v16bf a = cat8(qrow[hi], qrow[2 + hi]);
        for (int nt = 0; nt < 4; ++nt) {
            const v8bf* krow = (const v8bf*)(&sK[head][nt * 16 + n][0]);
            v16bf b = cat8(krow[hi], krow[2 + hi]);
            v8f c = {};
            c = __builtin_amdgcn_wmma_f32_16x16x32_bf16(false, a, false, b,
                                                        (short)0, c, false, false);
#pragma unroll
            for (int rr = 0; rr < 8; ++rr)
                sA[head][mt * 16 + mofs + rr][nt * 16 + n] = c[rr];
        }
    }
    __syncthreads();
    // phase 2: + rel-pos bias, 0.25*softmax + 0.75*relu^2  (one row per thread)
    {
        int h2 = tid >> 6, q = tid & 63, qi = q >> 3, qj = q & 7;
        float mx = -3.4e38f;
        for (int k = 0; k < 64; ++k) {
            int ki = k >> 3, kj = k & 7;
            float bv = rpb[((qi - ki + 7) * 15 + (qj - kj + 7)) * HEADS + h2];
            float v = sA[h2][q][k] + bv;
            sA[h2][q][k] = v;
            mx = fmaxf(mx, v);
        }
        float s = 0.f;
        for (int k = 0; k < 64; ++k) s += expf(sA[h2][q][k] - mx);
        float inv = 1.f / s;
        for (int k = 0; k < 64; ++k) {
            float v = sA[h2][q][k];
            float e = expf(v - mx) * inv;
            float rl = fmaxf(v, 0.f);
            sA[h2][q][k] = 0.25f * e + 0.75f * rl * rl;
        }
    }
    __syncthreads();
    // phase 3: out = attn * V  (K = 64 tokens -> 2 WMMA steps)
    for (int mt = (wave & 1) * 2; mt < (wave & 1) * 2 + 2; ++mt) {
        v8f c = {};
        const v8bf* vrow = (const v8bf*)(&sVt[head * HD + n][0]);
#pragma unroll
        for (int ks = 0; ks < 2; ++ks) {
            const float* arow = &sA[head][mt * 16 + m][ks * 32];
            v16bf a;
#pragma unroll
            for (int i = 0; i < 8; ++i) a[i] = f2bf(arow[kb8 + i]);
#pragma unroll
            for (int i = 0; i < 8; ++i) a[8 + i] = f2bf(arow[16 + kb8 + i]);
            v16bf b = cat8(vrow[4 * ks + hi], vrow[4 * ks + 2 + hi]);
            c = __builtin_amdgcn_wmma_f32_16x16x32_bf16(false, a, false, b,
                                                        (short)0, c, false, false);
        }
#pragma unroll
        for (int rr = 0; rr < 8; ++rr) {
            int tok = mt * 16 + mofs + rr;
            size_t o = ((size_t)bt * Cch + head * HD + n) * HW +
                       (size_t)(wh0 + (tok >> 3)) * Wimg + ww0 + (tok & 7);
            out[o] = c[rr];
        }
    }
}

// ------------------------------------------------- MDTA: row L2 norms (q & k)
__global__ __launch_bounds__(256) void mdta_norm_kernel(const float* __restrict__ qkv,
                                                        float* __restrict__ norms) {
    __shared__ float red[256];
    int rowid = blockIdx.x;                 // 0..1023  (q rows then k rows)
    int qk = rowid >> 9;
    int r = rowid & 511; int bt = r >> 6; int ch = r & 63;
    const float* base = qkv + ((size_t)bt * 192 + qk * 64 + ch) * HW;
    float s = 0.f;
    for (int i = threadIdx.x; i < HW; i += 256) { float v = base[i]; s += v * v; }
    red[threadIdx.x] = s;
    __syncthreads();
    for (int st = 128; st > 0; st >>= 1) {
        if ((int)threadIdx.x < st) red[threadIdx.x] += red[threadIdx.x + st];
        __syncthreads();
    }
    if (threadIdx.x == 0) norms[rowid] = fmaxf(sqrtf(red[0]), 1e-12f);
}

// ------------------- MDTA: 16x16 Gram matrix over HW=16384 + softmax (1 wave)
__global__ __launch_bounds__(32) void mdta_attn_kernel(const float* __restrict__ qkv,
                                                       const float* __restrict__ norms,
                                                       const float* __restrict__ temp,
                                                       float* __restrict__ attnM) {
    __shared__ float sAt[16][16];
    int bt = blockIdx.x >> 2, head = blockIdx.x & 3;
    int lane = threadIdx.x;
    int m = lane & 15, hi = (lane >> 4) & 1;
    int kb8 = hi * 8, kb16 = hi * 16, mofs = hi * 8, n = lane & 15;
    const float* qb  = qkv + ((size_t)bt * 192 + head * HD) * HW;
    const float* kbp = qkv + ((size_t)bt * 192 + 64 + head * HD) * HW;
    v8f c = {};
    for (int k0 = 0; k0 < HW; k0 += 32) {
        const float* qrow = qb + (size_t)m * HW + k0;
        v4f q0 = *(const v4f*)(qrow + kb8);
        v4f q1 = *(const v4f*)(qrow + kb8 + 4);
        v4f q2 = *(const v4f*)(qrow + 16 + kb8);
        v4f q3 = *(const v4f*)(qrow + 16 + kb8 + 4);
        const float* krow = kbp + (size_t)n * HW + k0 + kb8;
        v4f p0 = *(const v4f*)(krow);
        v4f p1 = *(const v4f*)(krow + 4);
        v4f p2 = *(const v4f*)(krow + 16);
        v4f p3 = *(const v4f*)(krow + 20);
        v16bf a, b;
#pragma unroll
        for (int i = 0; i < 4; ++i) {
            a[i] = f2bf(q0[i]); a[4 + i] = f2bf(q1[i]);
            a[8 + i] = f2bf(q2[i]); a[12 + i] = f2bf(q3[i]);
            b[i] = f2bf(p0[i]); b[4 + i] = f2bf(p1[i]);
            b[8 + i] = f2bf(p2[i]); b[12 + i] = f2bf(p3[i]);
        }
        c = __builtin_amdgcn_wmma_f32_16x16x32_bf16(false, a, false, b,
                                                    (short)0, c, false, false);
    }
#pragma unroll
    for (int rr = 0; rr < 8; ++rr) sAt[mofs + rr][n] = c[rr];
    __syncthreads();
    if (lane < 16) {
        float nq = norms[bt * 64 + head * HD + lane];
        float tp = temp[head];
        float v[16]; float mx = -3.4e38f;
        for (int d = 0; d < 16; ++d) {
            float nk = norms[512 + bt * 64 + head * HD + d];
            v[d] = sAt[lane][d] / (nq * nk) * tp;
            mx = fmaxf(mx, v[d]);
        }
        float s = 0.f;
        for (int d = 0; d < 16; ++d) { v[d] = expf(v[d] - mx); s += v[d]; }
        float inv = 1.f / s;
        float* ob = attnM + ((size_t)(bt * HEADS + head) * 16 + lane) * 16;
        for (int d = 0; d < 16; ++d) ob[d] = v[d] * inv;
    }
}

// ----------------------------- MDTA: out = attn(16x16) * V(16 x HW), per tile
__global__ __launch_bounds__(32) void mdta_out_kernel(const float* __restrict__ attnM,
                                                      const float* __restrict__ qkv,
                                                      float* __restrict__ out) {
    __shared__ __attribute__((aligned(32))) __bf16 sAp[16][32];  // K padded, zeros
    __shared__ __attribute__((aligned(32))) __bf16 sVt2[16][32]; // [pix][k] padded
    int g = blockIdx.x;
    int ptile = g & 1023; int head = (g >> 10) & 3; int bt = g >> 12;
    int pix0 = ptile << 4;
    int lane = threadIdx.x;
    int m = lane & 15, hi = (lane >> 4) & 1;
    int mofs = hi * 8, n = lane & 15;
    const float* ab = attnM + ((size_t)(bt * HEADS + head) * 16) * 16;
    const float* vb = qkv + ((size_t)bt * 192 + 128 + head * HD) * HW;
    for (int e = lane; e < 256; e += 32) {
        int r = e >> 4, ccol = e & 15;
        sAp[r][ccol]       = f2bf(ab[e]);
        sAp[r][16 + ccol]  = f2bf(0.f);
        sVt2[ccol][r]      = f2bf(vb[(size_t)r * HW + pix0 + ccol]);
        sVt2[ccol][16 + r] = f2bf(0.f);
    }
    __syncthreads();
    const v8bf* arow = (const v8bf*)(&sAp[m][0]);
    const v8bf* vrow = (const v8bf*)(&sVt2[n][0]);
    v16bf a = cat8(arow[hi], arow[2 + hi]);
    v16bf b = cat8(vrow[hi], vrow[2 + hi]);
    v8f c = {};
    c = __builtin_amdgcn_wmma_f32_16x16x32_bf16(false, a, false, b,
                                                (short)0, c, false, false);
#pragma unroll
    for (int rr = 0; rr < 8; ++rr)
        out[((size_t)bt * Cch + head * HD + mofs + rr) * HW + pix0 + n] = c[rr];
}

// --------------------- GDFN: fused dual depthwise 3x3 + exact GELU + gating
__global__ __launch_bounds__(256) void gdfn_dw_kernel(const float* __restrict__ in,
                                                      const float* __restrict__ w1,
                                                      const float* __restrict__ b1,
                                                      const float* __restrict__ w2,
                                                      const float* __restrict__ b2,
                                                      float* __restrict__ out) {
    size_t idx = (size_t)blockIdx.x * 256 + threadIdx.x;
    size_t total = (size_t)BT * HIDc * HW;
    if (idx >= total) return;
    int pos = (int)(idx % HW);
    size_t t1 = idx / HW;
    int c = (int)(t1 % HIDc); int bt = (int)(t1 / HIDc);
    int hh = pos >> 7, ww = pos & 127;
    const float* base1 = in + ((size_t)bt * (2 * HIDc) + c) * HW;
    const float* base2 = in + ((size_t)bt * (2 * HIDc) + HIDc + c) * HW;
    const float* wc1 = w1 + (size_t)c * 9;
    const float* wc2 = w2 + (size_t)c * 9;
    float a1 = b1[c], a2 = b2[c];
#pragma unroll
    for (int ky = 0; ky < 3; ++ky)
#pragma unroll
        for (int kx = 0; kx < 3; ++kx) {
            int y = hh + ky - 1, x = ww + kx - 1;
            if (y >= 0 && y < Himg && x >= 0 && x < Wimg) {
                a1 += wc1[ky * 3 + kx] * base1[y * Wimg + x];
                a2 += wc2[ky * 3 + kx] * base2[y * Wimg + x];
            }
        }
    float gl = 0.5f * a1 * (1.f + erff(a1 * 0.70710678118654752f));  // exact GELU
    out[idx] = gl * a2;
}

// ============================================================== host launcher
extern "C" void kernel_launch(void* const* d_in, const int* in_sizes, int n_in,
                              void* d_out, int out_size, void* d_ws, size_t ws_size,
                              hipStream_t stream) {
    const float* x        = (const float*)d_in[0];
    const float* norm_w   = (const float*)d_in[1];
    const float* norm_b   = (const float*)d_in[2];
    const float* s_cin_w  = (const float*)d_in[3];
    const float* s_cin_b  = (const float*)d_in[4];
    const float* s_dw_w   = (const float*)d_in[5];
    const float* s_dw_b   = (const float*)d_in[6];
    const float* s_rpb    = (const float*)d_in[7];
    const float* s_cout_w = (const float*)d_in[8];
    const float* s_cout_b = (const float*)d_in[9];
    const float* m_qkv_w  = (const float*)d_in[10];
    const float* m_dw_w   = (const float*)d_in[11];
    const float* m_proj_w = (const float*)d_in[12];
    const float* m_temp   = (const float*)d_in[13];
    const float* g_in_w   = (const float*)d_in[14];
    const float* g_in_b   = (const float*)d_in[15];
    const float* g_d1_w   = (const float*)d_in[16];
    const float* g_d1_b   = (const float*)d_in[17];
    const float* g_d2_w   = (const float*)d_in[18];
    const float* g_d2_b   = (const float*)d_in[19];
    const float* g_out_w  = (const float*)d_in[20];
    const float* g_out_b  = (const float*)d_in[21];
    float* out = (float*)d_out;

    float* ws      = (float*)d_ws;
    float* ws_ln   = ws;                       //  64 planes
    float* ws_A    = ws_ln + 64 * NP;          // 352 planes (max 340 ch)
    float* ws_B    = ws_A + 352 * NP;          // 192 planes
    float* ws_C    = ws_B + 192 * NP;          //  64 planes
    float* ws_nrm  = ws_C + 64 * NP;           // 1024 floats
    float* ws_attn = ws_nrm + 1024;            // 8192 floats
    __bf16* ws_wp  = (__bf16*)(ws_attn + 8192); // packed weights (<= 72KB)

    const int LN_GRID   = (int)(NP / 256);       // 512
    const int GEMM_GRID = (int)(NP / 64);        // 2048
    const int WIN_GRID  = BT * (Himg / WIN) * (Wimg / WIN); // 2048

    auto packGrid = [](int Cout, int Kpad) {
        int MT16 = ((Cout + 15) / 16) * 16;
        return (MT16 * Kpad + 255) / 256;
    };

    // ---- SWSA branch: x1 = x + swsa(LN(x))
    ln_kernel<<<LN_GRID, 256, 0, stream>>>(x, norm_w, norm_b, ws_ln);
    pack_w_kernel<<<packGrid(192, 192), 256, 0, stream>>>(s_cin_w, ws_wp, 192, 192, 192);
    gemm_conv_kernel<true><<<GEMM_GRID, 256, 0, stream>>>(
        ws_ln, ws_wp, s_cin_b, nullptr, ws_A, 64, 192, 192, 192);
    dw_kernel<<<(int)((size_t)BT * 192 * HW / 256), 256, 0, stream>>>(
        ws_A, s_dw_w, s_dw_b, ws_B, 192, 1);
    swsa_attn_kernel<<<WIN_GRID, 256, 0, stream>>>(ws_B, s_rpb, ws_C);
    pack_w_kernel<<<packGrid(64, 192), 256, 0, stream>>>(s_cout_w, ws_wp, 64, 192, 192);
    gemm_conv_kernel<true><<<GEMM_GRID, 256, 0, stream>>>(
        ws_C, ws_wp, s_cout_b, x, out, 64, 64, 192, 192);

    // ---- MDTA branch: x2 = x1 + mdta(LN(x1))
    ln_kernel<<<LN_GRID, 256, 0, stream>>>(out, norm_w, norm_b, ws_ln);
    pack_w_kernel<<<packGrid(192, 64), 256, 0, stream>>>(m_qkv_w, ws_wp, 192, 64, 64);
    gemm_conv_kernel<false><<<GEMM_GRID, 256, 0, stream>>>(
        ws_ln, ws_wp, nullptr, nullptr, ws_A, 64, 192, 64, 64);
    dw_kernel<<<(int)((size_t)BT * 192 * HW / 256), 256, 0, stream>>>(
        ws_A, m_dw_w, nullptr, ws_B, 192, 2);
    mdta_norm_kernel<<<1024, 256, 0, stream>>>(ws_B, ws_nrm);
    mdta_attn_kernel<<<BT * HEADS, 32, 0, stream>>>(ws_B, ws_nrm, m_temp, ws_attn);
    mdta_out_kernel<<<BT * HEADS * (HW / 16), 32, 0, stream>>>(ws_attn, ws_B, ws_C);
    pack_w_kernel<<<packGrid(64, 64), 256, 0, stream>>>(m_proj_w, ws_wp, 64, 64, 64);
    gemm_conv_kernel<false><<<GEMM_GRID, 256, 0, stream>>>(
        ws_C, ws_wp, nullptr, out, out, 64, 64, 64, 64);

    // ---- GDFN branch: x3 = x2 + gdfn(LN(x2))
    ln_kernel<<<LN_GRID, 256, 0, stream>>>(out, norm_w, norm_b, ws_ln);
    pack_w_kernel<<<packGrid(2 * HIDc, 64), 256, 0, stream>>>(g_in_w, ws_wp, 2 * HIDc, 64, 64);
    gemm_conv_kernel<false><<<GEMM_GRID, 256, 0, stream>>>(
        ws_ln, ws_wp, g_in_b, nullptr, ws_A, 64, 2 * HIDc, 64, 64);
    gdfn_dw_kernel<<<(int)((size_t)BT * HIDc * HW / 256), 256, 0, stream>>>(
        ws_A, g_d1_w, g_d1_b, g_d2_w, g_d2_b, ws_B);
    pack_w_kernel<<<packGrid(64, 192), 256, 0, stream>>>(g_out_w, ws_wp, 64, HIDc, 192);
    gemm_conv_kernel<false><<<GEMM_GRID, 256, 0, stream>>>(
        ws_B, ws_wp, g_out_b, out, out, HIDc, 64, HIDc, 192);
}
